// cVAE_65420941853138
// MI455X (gfx1250) — matrix operand order repaired and checked by reading, not compiled, not run on hardware
//
#include <hip/hip_runtime.h>
#include <hip/hip_bf16.h>
#include <math.h>

// ---------------------------------------------------------------------------
// cVAE + mixture-of-experts decoder for MI455X (gfx1250, wave32, WMMA).
// Compute-bound (~314 GFLOP; HBM traffic ~10us at 23.3 TB/s), so all GEMMs
// run on v_wmma_f32_16x16x32_f16 with f32 accumulation.
//
// Weights are pre-packed ONCE into the WMMA B-fragment layout
//   frag(kt, nf): [lane 0..31][16 contiguous halves]
//   lane = g*16+ln -> column nf*16+ln, halves j -> K = kt*32 + g*16 + j
// so B tiles stream into LDS as b128 copies and fragments read back as two
// contiguous ds_load_b128 per lane (no scalar gather, no packing movs).
//
// Staging uses gfx1250 GLOBAL_LOAD_ASYNC_TO_LDS_B128 (ASYNCcnt) with a
// double-buffered pipeline: tile kt+1 streams to LDS while WMMA consumes
// tile kt.  Falls back to global_load_b128 + ds_store_b128 if the async
// builtin is unavailable on the toolchain.
//
// MoE fusion: out = sum_e (c_e * inp) @ W_e + sum_e c_e*bias_e -- coefficient
// folded into the A fragment (one pk_mul chain per expert), single
// accumulator set across experts and K.
// ---------------------------------------------------------------------------

typedef _Float16 f16;
typedef __attribute__((ext_vector_type(16))) _Float16 v16h;
typedef __attribute__((ext_vector_type(8)))  _Float16 v8h;
typedef __attribute__((ext_vector_type(8)))  float    v8f;
typedef int i32x4 __attribute__((vector_size(16)));   // builtin's b128 unit

#define BM 256   // 8 waves x 32 rows
#define BN 64    // 4 x 16-wide WMMA fragments per wave
#define BK 32    // K step of v_wmma_f32_16x16x32_f16

#if defined(__gfx1250__) && __has_builtin(__builtin_amdgcn_global_load_async_to_lds_b128)
#define USE_ASYNC_LDS 1
#else
#define USE_ASYNC_LDS 0
#endif

__device__ __forceinline__ void wait_async_zero() {
#if __has_builtin(__builtin_amdgcn_s_wait_asynccnt)
  __builtin_amdgcn_s_wait_asynccnt(0);
#else
  asm volatile("s_wait_asynccnt 0x0" ::: "memory");
#endif
}

// ---- elementwise helpers ---------------------------------------------------

// Pack f32 weights [nexp][K][N] into WMMA fragment layout:
// dst[((e*KT + kt)*NFP + nf)*512 + lane*16 + j] = W_e[kt*32 + (lane>>4)*16 + j][nf*16 + (lane&15)]
// (zero-filled outside K x N; NFP is padded to a multiple of 4 fragments)
__global__ void pack_weights(const float* __restrict__ src, f16* __restrict__ dst,
                             int K, int N, int KT, int NFP, int nexp) {
  long i = (long)blockIdx.x * blockDim.x + threadIdx.x;
  long per = (long)KT * NFP * 512;
  if (i >= per * nexp) return;
  int  e  = (int)(i / per);
  long r  = i - (long)e * per;
  int  kt = (int)(r / ((long)NFP * 512));
  long r2 = r - (long)kt * NFP * 512;
  int  nf   = (int)(r2 >> 9);
  int  lj   = (int)(r2 & 511);
  int  lane = lj >> 4;
  int  j    = lj & 15;
  int  k = kt * 32 + (lane >> 4) * 16 + j;
  int  n = nf * 16 + (lane & 15);
  f16 v = (f16)0;
  if (k < K && n < N) v = (f16)src[((long)e * K + k) * N + n];
  dst[i] = v;
}

// dst[r*dld + c] = (f16) src[r*sld + c]   (dst pointer already column-offset)
__global__ void copy_cast2d(const float* __restrict__ src, int sld,
                            f16* __restrict__ dst, int dld, int rows, int cols) {
  int i = blockIdx.x * blockDim.x + threadIdx.x;
  if (i >= rows * cols) return;
  int r = i / cols, c = i - r * cols;
  dst[(long)r * dld + c] = (f16)src[(long)r * sld + c];
}

// z = mu + eps * exp(0.5*logvar); written (f16) into cols [0,LAT) of two buffers
__global__ void reparam_kernel(const float* __restrict__ mu, const float* __restrict__ lv,
                               const float* __restrict__ eps,
                               f16* __restrict__ dA, f16* __restrict__ dB,
                               int n, int lat, int ld) {
  int i = blockIdx.x * blockDim.x + threadIdx.x;
  if (i >= n) return;
  float z = mu[i] + eps[i] * __expf(0.5f * lv[i]);
  int r = i / lat, c = i - r * lat;
  long o = (long)r * ld + c;
  dA[o] = (f16)z;
  dB[o] = (f16)z;
}

__global__ void softmax_rows(const float* __restrict__ in, float* __restrict__ out,
                             int rows, int cols) {
  int r = blockIdx.x * blockDim.x + threadIdx.x;
  if (r >= rows) return;
  float m = -INFINITY;
  for (int c = 0; c < cols; ++c) m = fmaxf(m, in[(long)r * cols + c]);
  float e[8];   // cols <= 8
  float s = 0.0f;
  for (int c = 0; c < cols; ++c) { e[c] = __expf(in[(long)r * cols + c] - m); s += e[c]; }
  float inv = 1.0f / s;
  for (int c = 0; c < cols; ++c) out[(long)r * cols + c] = e[c] * inv;
}

// ---- WMMA GEMM with fused expert mixture ----------------------------------
// NEXP==1: D = A @ W + bias.  NEXP>1: D = sum_e (c_e*A) @ W_e + sum_e c_e*b_e.
// A: [8192, K] f16 row-major (lda); Wp: packed fragments (see pack_weights).
template <int NEXP>
__global__ __launch_bounds__(256) void wmma_gemm_moe(
    const f16* __restrict__ A, int lda,
    const f16* __restrict__ Wp, long long wstride, int NFP,
    const float* __restrict__ bias, int bstride,
    const float* __restrict__ coeff,
    int N, int K,
    float* __restrict__ outF32, int ldf,
    f16* __restrict__ outH, int ldh,
    int do_elu) {
  // double-buffered tiles: [buf][expert][frag 0..3][lane][16 halves]
  __shared__ __align__(16) f16 Ls[2][NEXP * 2048];

  const int tid    = threadIdx.x;
  const int lane   = tid & 31;
  const int wave   = tid >> 5;
  const int mwave  = blockIdx.x * BM + wave * 32;   // 32 rows per wave
  const int nfrag0 = blockIdx.y * 4;                // 4 n-fragments per block
  const int g  = lane >> 4;
  const int ln = lane & 15;
  const int KT = K >> 5;

  // Stage one 32x64 packed tile (4 contiguous frag-blocks per expert = 4KB)
  // into LDS buffer `buf`: 16B per thread per expert.
  auto stage = [&](int kt, int buf) {
#pragma unroll
    for (int e = 0; e < NEXP; ++e) {
      const f16* src = Wp + (long long)e * wstride
                          + ((long)kt * NFP + nfrag0) * 512 + tid * 8;
      f16* dst = &Ls[buf][e * 2048 + tid * 8];
#if USE_ASYNC_LDS
      __builtin_amdgcn_global_load_async_to_lds_b128(
          (__attribute__((address_space(1))) i32x4*)src,
          (__attribute__((address_space(3))) i32x4*)dst, 0, 0);
#else
      *(v8h*)dst = *(const v8h*)src;
#endif
    }
  };

  // f16 coefficient scales for this lane's two A rows (registers: loops unroll)
  f16 cs0[NEXP], cs1[NEXP];
#pragma unroll
  for (int e = 0; e < NEXP; ++e) {
    if (NEXP > 1) {
      cs0[e] = (f16)coeff[(long)(mwave + ln) * NEXP + e];
      cs1[e] = (f16)coeff[(long)(mwave + 16 + ln) * NEXP + e];
    } else {
      cs0[e] = (f16)1.0f; cs1[e] = (f16)1.0f;
    }
  }

  v8f acc[8] = {};   // [mfrag 0..1][nfrag 0..3], accumulated over experts AND K

  stage(0, 0);
  for (int kt = 0; kt < KT; ++kt) {
#if USE_ASYNC_LDS
    wait_async_zero();        // own tile-kt asyncs done ...
#endif
    __syncthreads();          // ... and everyone's (also: compute(kt-1) done)
    if (kt + 1 < KT) stage(kt + 1, (kt + 1) & 1);  // overlaps with compute(kt)

    // A fragments (contiguous k order, same as packed B): halves [0..7] = K k0+g*16..,
    // halves [8..15] = K k0+g*16+8.. ; rows mwave+ln and mwave+16+ln.
    const int k0 = kt * 32;
    const f16* Ap0 = A + (long)(mwave + ln) * lda + k0 + g * 16;
    const f16* Ap1 = Ap0 + (long)16 * lda;
    v16h a0, a1;
    {
      v8h l0 = *(const v8h*)Ap0, h0 = *(const v8h*)(Ap0 + 8);
      v8h l1 = *(const v8h*)Ap1, h1 = *(const v8h*)(Ap1 + 8);
#pragma unroll
      for (int j = 0; j < 8; ++j) {
        a0[j] = l0[j]; a0[8 + j] = h0[j];
        a1[j] = l1[j]; a1[8 + j] = h1[j];
      }
    }

    const f16* lbuf = Ls[kt & 1];
#pragma unroll
    for (int e = 0; e < NEXP; ++e) {
      v16h a0s = a0, a1s = a1;
      if (NEXP > 1) {
#pragma unroll
        for (int j = 0; j < 16; ++j) { a0s[j] *= cs0[e]; a1s[j] *= cs1[e]; }
      }
      const f16* lsb = lbuf + e * 2048;
#pragma unroll
      for (int f = 0; f < 4; ++f) {
        // B fragment: 32 contiguous bytes per lane -> two ds_load_b128
        const v8h* bp = (const v8h*)(lsb + (f * 32 + lane) * 16);
        v8h blo = bp[0], bhi = bp[1];
        v16h bf;
#pragma unroll
        for (int j = 0; j < 8; ++j) { bf[j] = blo[j]; bf[8 + j] = bhi[j]; }
        acc[f]     = __builtin_amdgcn_wmma_f32_16x16x32_f16(
            false, a0s, false, bf, (short)0, acc[f],     false, false);
        acc[4 + f] = __builtin_amdgcn_wmma_f32_16x16x32_f16(
            false, a1s, false, bf, (short)0, acc[4 + f], false, false);
      }
    }
  }

  // Epilogue: mixed bias, optional ELU, optional f32 / f16 stores.
  // C/D layout: VGPR r holds row (g*8 + r), column = lane&15.
#pragma unroll
  for (int mf = 0; mf < 2; ++mf) {
#pragma unroll
    for (int f = 0; f < 4; ++f) {
      int ng = (nfrag0 + f) * 16 + ln;
      if (ng >= N) continue;
#pragma unroll
      for (int r = 0; r < 8; ++r) {
        int mg = mwave + mf * 16 + g * 8 + r;
        float v = acc[mf * 4 + f][r];
        if (NEXP > 1) {
          float bm = 0.0f;
#pragma unroll
          for (int e = 0; e < NEXP; ++e)
            bm += coeff[(long)mg * NEXP + e] * bias[(long)e * bstride + ng];
          v += bm;
        } else {
          v += bias[ng];
        }
        if (do_elu) v = (v > 0.0f) ? v : expm1f(v);
        if (outF32) outF32[(long)mg * ldf + ng] = v;
        if (outH)   outH[(long)mg * ldh + ng] = (f16)v;
      }
    }
  }
}

// ---- host side -------------------------------------------------------------

static void launch_gemm(hipStream_t s, const f16* A, int lda, const f16* Wp,
                        int K, int N,
                        const float* bias, int bstride,
                        const float* coeff, int nexp,
                        float* of, int ldf, f16* oh, int ldh, int elu) {
  int NFP = 4 * ((N + 63) / 64);
  long long wstride = (long long)(K / 32) * NFP * 512;
  dim3 grid(8192 / BM, (N + BN - 1) / BN);
  if (nexp == 1)
    hipLaunchKernelGGL((wmma_gemm_moe<1>), grid, dim3(256), 0, s,
                       A, lda, Wp, wstride, NFP, bias, bstride, coeff,
                       N, K, of, ldf, oh, ldh, elu);
  else
    hipLaunchKernelGGL((wmma_gemm_moe<6>), grid, dim3(256), 0, s,
                       A, lda, Wp, wstride, NFP, bias, bstride, coeff,
                       N, K, of, ldf, oh, ldh, elu);
}

extern "C" void kernel_launch(void* const* d_in, const int* in_sizes, int n_in,
                              void* d_out, int out_size, void* d_ws, size_t ws_size,
                              hipStream_t stream) {
  const int B_ = 8192, IN_ = 256, COND_ = 1024, H_ = 1024, LAT_ = 128, E_ = 6, GH_ = 64;
  const int INP_ = LAT_ + COND_;          // 1152 (decoder/gating input width)
  const int K0 = IN_ + COND_ + IN_;       // 1536
  const int K1 = H_ + IN_;                // 1280

  const float* x      = (const float*)d_in[0];
  const float* y      = (const float*)d_in[1];
  const float* eps    = (const float*)d_in[2];
  const float* enc_w0 = (const float*)d_in[3];
  const float* enc_b0 = (const float*)d_in[4];
  const float* enc_w1 = (const float*)d_in[5];
  const float* enc_b1 = (const float*)d_in[6];
  const float* mu_w   = (const float*)d_in[7];
  const float* mu_b   = (const float*)d_in[8];
  const float* lv_w   = (const float*)d_in[9];
  const float* lv_b   = (const float*)d_in[10];
  const float* g_w0   = (const float*)d_in[11];
  const float* g_b0   = (const float*)d_in[12];
  const float* g_w1   = (const float*)d_in[13];
  const float* g_b1   = (const float*)d_in[14];
  const float* g_w2   = (const float*)d_in[15];
  const float* g_b2   = (const float*)d_in[16];
  const float* exp_w0 = (const float*)d_in[17];
  const float* exp_b0 = (const float*)d_in[18];
  const float* exp_w1 = (const float*)d_in[19];
  const float* exp_b1 = (const float*)d_in[20];
  const float* exp_w2 = (const float*)d_in[21];
  const float* exp_b2 = (const float*)d_in[22];

  float* out    = (float*)d_out;                 // [8192,256] layer_out
  float* out_mu = out + (long)B_ * IN_;          // [8192,128]
  float* out_lv = out_mu + (long)B_ * LAT_;      // [8192,128]

  // 256B-aligned workspace carve-out (keeps 16B vector loads legal).
  char* wsb = (char*)d_ws;
  size_t off = 0;
  auto alloc = [&](size_t bytes) -> void* {
    off = (off + 255) & ~(size_t)255;
    void* p = wsb + off;
    off += bytes;
    return p;
  };
  auto packed_elems = [](int K, int N, int nexp) -> size_t {
    int NFP = 4 * ((N + 63) / 64);
    return (size_t)nexp * (K / 32) * NFP * 512;
  };

  f16* enc_w0p = (f16*)alloc(packed_elems(K0, H_, 1) * 2);
  f16* enc_w1p = (f16*)alloc(packed_elems(K1, H_, 1) * 2);
  f16* mu_wp   = (f16*)alloc(packed_elems(H_, LAT_, 1) * 2);
  f16* lv_wp   = (f16*)alloc(packed_elems(H_, LAT_, 1) * 2);
  f16* g_w0p   = (f16*)alloc(packed_elems(INP_, GH_, 1) * 2);
  f16* g_w1p   = (f16*)alloc(packed_elems(GH_, GH_, 1) * 2);
  f16* g_w2p   = (f16*)alloc(packed_elems(GH_, E_, 1) * 2);
  f16* exp_w0p = (f16*)alloc(packed_elems(INP_, H_, E_) * 2);
  f16* exp_w1p = (f16*)alloc(packed_elems(INP_, H_, E_) * 2);
  f16* exp_w2p = (f16*)alloc(packed_elems(INP_, IN_, E_) * 2);
  f16* enc_in0 = (f16*)alloc((size_t)B_ * K0 * 2);     // [x, y, x]
  f16* enc_in1 = (f16*)alloc((size_t)B_ * K1 * 2);     // [h0, x]
  f16* h2      = (f16*)alloc((size_t)B_ * H_ * 2);     // encoder layer1 out
  f16* bufA    = (f16*)alloc((size_t)B_ * INP_ * 2);   // [z, y] then [z, out1]
  f16* bufB    = (f16*)alloc((size_t)B_ * INP_ * 2);   // [z, out0]
  f16* gbuf0   = (f16*)alloc((size_t)B_ * GH_ * 2);
  f16* gbuf1   = (f16*)alloc((size_t)B_ * GH_ * 2);
  float* logits = (float*)alloc((size_t)B_ * E_ * 4);
  float* coeff  = (float*)alloc((size_t)B_ * E_ * 4);

  auto pack = [&](const float* s, f16* d, int K, int N, int nexp) {
    long n = (long)packed_elems(K, N, nexp);
    int NFP = 4 * ((N + 63) / 64);
    hipLaunchKernelGGL(pack_weights, dim3((n + 255) / 256), dim3(256), 0, stream,
                       s, d, K, N, K / 32, NFP, nexp);
  };
  auto cc2d = [&](const float* s, int sld, f16* d, int dld, int rows, int cols) {
    long n = (long)rows * cols;
    hipLaunchKernelGGL(copy_cast2d, dim3((n + 255) / 256), dim3(256), 0, stream,
                       s, sld, d, dld, rows, cols);
  };

  // 1) Weights -> packed f16 WMMA fragments
  pack(enc_w0, enc_w0p, K0, H_, 1);
  pack(enc_w1, enc_w1p, K1, H_, 1);
  pack(mu_w,   mu_wp,   H_, LAT_, 1);
  pack(lv_w,   lv_wp,   H_, LAT_, 1);
  pack(g_w0,   g_w0p,   INP_, GH_, 1);
  pack(g_w1,   g_w1p,   GH_, GH_, 1);
  pack(g_w2,   g_w2p,   GH_, E_, 1);
  pack(exp_w0, exp_w0p, INP_, H_, E_);
  pack(exp_w1, exp_w1p, INP_, H_, E_);
  pack(exp_w2, exp_w2p, INP_, IN_, E_);

  // 2) Concats as column-offset f16 writes
  cc2d(x, IN_,   enc_in0,               K0,  B_, IN_);    // [:,0:256)    = x
  cc2d(y, COND_, enc_in0 + IN_,         K0,  B_, COND_);  // [:,256:1280) = y
  cc2d(x, IN_,   enc_in0 + IN_ + COND_, K0,  B_, IN_);    // [:,1280:1536)= x
  cc2d(x, IN_,   enc_in1 + H_,          K1,  B_, IN_);    // enc layer1 tail
  cc2d(y, COND_, bufA + LAT_,           INP_, B_, COND_); // [z|y] tail

  // 3) Encoder
  launch_gemm(stream, enc_in0, K0, enc_w0p, K0, H_, enc_b0, 0, nullptr, 1,
              nullptr, 0, enc_in1, K1, 1);                 // h0 (ELU) -> enc_in1[:,0:1024)
  launch_gemm(stream, enc_in1, K1, enc_w1p, K1, H_, enc_b1, 0, nullptr, 1,
              nullptr, 0, h2, H_, 1);                      // h1 (ELU)
  launch_gemm(stream, h2, H_, mu_wp, H_, LAT_, mu_b, 0, nullptr, 1,
              out_mu, LAT_, nullptr, 0, 0);                // mu -> d_out
  launch_gemm(stream, h2, H_, lv_wp, H_, LAT_, lv_b, 0, nullptr, 1,
              out_lv, LAT_, nullptr, 0, 0);                // logvar -> d_out

  // 4) Reparameterize: z into heads of bufA and bufB
  {
    long n = (long)B_ * LAT_;
    hipLaunchKernelGGL(reparam_kernel, dim3((n + 255) / 256), dim3(256), 0, stream,
                       out_mu, out_lv, eps, bufA, bufB, (int)n, LAT_, INP_);
  }

  // 5) Gating MLP + softmax
  launch_gemm(stream, bufA, INP_, g_w0p, INP_, GH_, g_b0, 0, nullptr, 1,
              nullptr, 0, gbuf0, GH_, 1);
  launch_gemm(stream, gbuf0, GH_, g_w1p, GH_, GH_, g_b1, 0, nullptr, 1,
              nullptr, 0, gbuf1, GH_, 1);
  launch_gemm(stream, gbuf1, GH_, g_w2p, GH_, E_, g_b2, 0, nullptr, 1,
              logits, E_, nullptr, 0, 0);
  hipLaunchKernelGGL(softmax_rows, dim3((B_ + 255) / 256), dim3(256), 0, stream,
                     logits, coeff, B_, E_);

  // 6) MoE decoder (coefficients folded into A; single accumulator set)
  launch_gemm(stream, bufA, INP_, exp_w0p, INP_, H_, exp_b0, H_, coeff, E_,
              nullptr, 0, bufB + LAT_, INP_, 1);           // layer0 -> bufB tail
  launch_gemm(stream, bufB, INP_, exp_w1p, INP_, H_, exp_b1, H_, coeff, E_,
              nullptr, 0, bufA + LAT_, INP_, 1);           // layer1 -> bufA tail
  launch_gemm(stream, bufA, INP_, exp_w2p, INP_, IN_, exp_b2, IN_, coeff, E_,
              out, IN_, nullptr, 0, 0);                    // layer2 -> d_out
  (void)n_in; (void)in_sizes; (void)out_size; (void)ws_size;
}
